// CausalSelfAttention_79113297592768
// MI455X (gfx1250) — matrix operand order repaired and checked by reading, not compiled, not run on hardware
//
#include <hip/hip_runtime.h>
#include <hip/hip_bf16.h>

typedef __attribute__((ext_vector_type(16))) __bf16 v16bf;
typedef __attribute__((ext_vector_type(8)))  __bf16 bf16x8;
typedef __attribute__((ext_vector_type(8)))  float  v8f;
typedef __attribute__((ext_vector_type(4)))  unsigned int u32x4;
typedef __attribute__((ext_vector_type(8)))  int i32x8;
typedef __attribute__((ext_vector_type(4)))  int i32x4;

__device__ __forceinline__ unsigned short f32_to_bf16(float f) {
    unsigned int u = __float_as_uint(f);
    unsigned int r = (u + 0x7FFFu + ((u >> 16) & 1u)) >> 16;   // RNE
    return (unsigned short)r;
}

// ---------------------------------------------------------------------------
// TDM: issue a 2D tile load (bf16 elements) global -> LDS. Wave-level op.
// D# layout per CDNA5 ISA ch.8: group0 = {count, lds_addr, global_addr, type},
// group1 = {mask/data_size, tensor_dim0/1, tile_dim0/1/2, strides}.
// ---------------------------------------------------------------------------
__device__ __forceinline__ void tdm_load_tile_2d(
    unsigned lds_addr,                 // LDS byte address of tile dst
    const unsigned short* gptr,        // global address of tile start
    unsigned tensor_d0,                // row length (elements)
    unsigned tensor_d1,                // number of rows (elements)
    unsigned long long stride0,        // row stride (elements)
    unsigned tile0, unsigned tile1)    // tile cols, tile rows
{
    unsigned long long ga = (unsigned long long)(size_t)gptr;
    u32x4 g0;
    g0[0] = 1u;                                        // count=1 (valid)
    g0[1] = lds_addr;                                  // [63:32] lds_addr
    g0[2] = (unsigned)ga;                              // global_addr lo
    g0[3] = (unsigned)((ga >> 32) & 0x01FFFFFFull)     // global_addr hi (57b)
          | (2u << 30);                                // type = 2 ("image")
    union { unsigned u[8]; i32x8 v; } g1;
    g1.u[0] = 0x00010000u;                             // wg_mask=0, data_size=1 (2B)
    g1.u[1] = (tensor_d0 & 0xFFFFu) << 16;             // tensor_dim0[15:0] @ bit48
    g1.u[2] = (tensor_d0 >> 16) | ((tensor_d1 & 0xFFFFu) << 16);
    g1.u[3] = (tensor_d1 >> 16) | (tile0 << 16);       // tile_dim0 @ bit112
    g1.u[4] = tile1;                                   // tile_dim1; tile_dim2=0
    g1.u[5] = (unsigned)(stride0 & 0xFFFFFFFFull);     // tensor_dim0_stride lo
    g1.u[6] = (unsigned)((stride0 >> 32) & 0xFFFFull); // stride0 hi; stride1=0
    g1.u[7] = 0;
    i32x4 z4 = {};
#if __clang_major__ >= 23
    i32x8 z8 = {};
    __builtin_amdgcn_tensor_load_to_lds(g0, g1.v, z4, z4, z8, 0);
#else
    __builtin_amdgcn_tensor_load_to_lds(g0, g1.v, z4, z4, 0);
#endif
}

// ---------------------------------------------------------------------------
// fp32 -> bf16 elementwise convert
// ---------------------------------------------------------------------------
__global__ __launch_bounds__(256) void cvt_f32_bf16(const float* __restrict__ in,
                                                    unsigned short* __restrict__ out,
                                                    int n) {
    int i = blockIdx.x * blockDim.x + threadIdx.x;
    int stride = gridDim.x * blockDim.x;
    for (; i < n; i += stride) out[i] = f32_to_bf16(in[i]);
}

// fp32 [K][N] -> bf16 [N][K] transpose-convert (for weight matrices)
__global__ __launch_bounds__(256) void tcvt_f32_bf16(const float* __restrict__ in,
                                                     unsigned short* __restrict__ out,
                                                     int K, int N) {
    int idx = blockIdx.x * blockDim.x + threadIdx.x;
    int stride = gridDim.x * blockDim.x;
    int total = K * N;
    for (; idx < total; idx += stride) {
        int k = idx / N, n = idx - k * N;
        out[(size_t)n * K + k] = f32_to_bf16(in[idx]);
    }
}

// ---------------------------------------------------------------------------
// Tiled WMMA GEMM with TDM-staged, double-buffered LDS tiles:
//   C[M][N] = A[M][K](bf16) * BT[N][K](bf16)^T + bias
// Block tile 128x128, 8 waves, wave tile 32x64 (2x4 wmma accumulators), BK=32.
// Wave 0 drives the Tensor Data Mover; TENSORcnt pipelines DMA vs WMMA.
// ---------------------------------------------------------------------------
template <bool BF16_OUT>
__global__ __launch_bounds__(256) void gemm_wmma(
    const unsigned short* __restrict__ A,   // [M][K] bf16
    const unsigned short* __restrict__ BT,  // [N][K] bf16 (B transposed)
    const float* __restrict__ bias,         // [N] fp32
    void* __restrict__ Cout,                // bf16 or f32 [M][N]
    int M, int N, int K)
{
    constexpr int BM = 128, BN = 128, BK = 32;
    __shared__ __align__(32) unsigned short sA[2][BM][BK];
    __shared__ __align__(32) unsigned short sB[2][BN][BK];

    const int tid  = threadIdx.x;
    const int lane = tid & 31;
    const int wave = tid >> 5;
    const int wr   = wave >> 1;        // 0..3 (M dir, 32 rows)
    const int wc   = wave & 1;         // 0..1 (N dir, 64 cols)
    const int lrow  = lane & 15;
    const int lhalf = lane >> 4;       // 0/1
    const int kh8   = lhalf * 8;       // A-frag K base within 32-chunk

    const int m0 = blockIdx.y * BM;
    const int n0 = blockIdx.x * BN;

    v8f acc[2][4] = {};

    const int niter = K / BK;
    if (wave == 0) {
        tdm_load_tile_2d((unsigned)(size_t)&sA[0][0][0], A + (size_t)m0 * K,
                         (unsigned)K, (unsigned)M, (unsigned long long)K, BK, BM);
        tdm_load_tile_2d((unsigned)(size_t)&sB[0][0][0], BT + (size_t)n0 * K,
                         (unsigned)K, (unsigned)N, (unsigned long long)K, BK, BN);
    }

    for (int it = 0; it < niter; ++it) {
        const int buf = it & 1;
        if (wave == 0) {
            if (it + 1 < niter) {
                const int k1 = (it + 1) * BK;
                tdm_load_tile_2d((unsigned)(size_t)&sA[buf ^ 1][0][0],
                                 A + (size_t)m0 * K + k1,
                                 (unsigned)K, (unsigned)M, (unsigned long long)K, BK, BM);
                tdm_load_tile_2d((unsigned)(size_t)&sB[buf ^ 1][0][0],
                                 BT + (size_t)n0 * K + k1,
                                 (unsigned)K, (unsigned)N, (unsigned long long)K, BK, BN);
                __builtin_amdgcn_s_wait_tensorcnt(2);  // current tile's 2 DMAs done
            } else {
                __builtin_amdgcn_s_wait_tensorcnt(0);
            }
        }
        __syncthreads();   // tile `buf` visible to all waves

        v16bf af[2], bfv[4];
#pragma unroll
        for (int mi = 0; mi < 2; ++mi) {
            int r = wr * 32 + mi * 16 + lrow;
            union { v16bf v; bf16x8 h[2]; } u;
            u.h[0] = *(const bf16x8*)&sA[buf][r][kh8];
            u.h[1] = *(const bf16x8*)&sA[buf][r][kh8 + 16];
            af[mi] = u.v;
        }
#pragma unroll
        for (int ni = 0; ni < 4; ++ni) {
            int r = wc * 64 + ni * 16 + lrow;
            union { v16bf v; bf16x8 h[2]; } ub;
            ub.h[0] = *(const bf16x8*)&sB[buf][r][kh8];
            ub.h[1] = *(const bf16x8*)&sB[buf][r][kh8 + 16];
            bfv[ni] = ub.v;
        }
#pragma unroll
        for (int mi = 0; mi < 2; ++mi)
#pragma unroll
            for (int ni = 0; ni < 4; ++ni)
                acc[mi][ni] = __builtin_amdgcn_wmma_f32_16x16x32_bf16(
                    false, af[mi], false, bfv[ni], (short)0, acc[mi][ni], false, false);

        __syncthreads();   // all waves done with `buf` before TDM refills it
    }

    // epilogue: C-layout — VGPR i holds row (i + 8*lhalf), lane%16 is column
#pragma unroll
    for (int mi = 0; mi < 2; ++mi) {
#pragma unroll
        for (int ni = 0; ni < 4; ++ni) {
            int col = n0 + wc * 64 + ni * 16 + lrow;
            float bv = bias[col];
#pragma unroll
            for (int i = 0; i < 8; ++i) {
                int row = m0 + wr * 32 + mi * 16 + i + lhalf * 8;
                float v = acc[mi][ni][i] + bv;
                if (BF16_OUT)
                    ((unsigned short*)Cout)[(size_t)row * N + col] = f32_to_bf16(v);
                else
                    ((float*)Cout)[(size_t)row * N + col] = v;
            }
        }
    }
}

// ---------------------------------------------------------------------------
// Flash attention (causal), per (b,h). Workgroup: 128 query rows, 8 waves of
// 16 queries each. K/V tiles of 32 keys staged in LDS per iteration.
// ---------------------------------------------------------------------------
__global__ __launch_bounds__(256) void attn_wmma(
    const unsigned short* __restrict__ Q,   // [B*T][1024] bf16
    const unsigned short* __restrict__ Kb,  // [B*T][1024] bf16
    const unsigned short* __restrict__ Vb,  // [B*T][1024] bf16
    unsigned short* __restrict__ O,         // [B*T][1024] bf16
    int T)
{
    const int bh   = blockIdx.y;           // 0..63
    const int b    = bh >> 4;
    const int h    = bh & 15;
    const int qblk = blockIdx.x;           // 0..15 -> 128 rows each
    const int tid  = threadIdx.x;
    const int lane = tid & 31;
    const int wave = tid >> 5;
    const int lrow  = lane & 15;
    const int lhalf = lane >> 4;

    const size_t headoff = (size_t)b * T * 1024 + (size_t)h * 64;

    __shared__ __align__(32) unsigned short sK[32][64];   // [key][d]
    __shared__ __align__(32) unsigned short sVT[64][32];  // [d][key]
    __shared__ __align__(32) unsigned short sP[8][16][32];

    const int qbase = qblk * 128 + wave * 16;

    // Q A-fragments for the two 32-wide contraction chunks of hd=64
    v16bf qf[2];
    {
        const unsigned short* qrow = Q + headoff + (size_t)(qbase + lrow) * 1024;
#pragma unroll
        for (int s = 0; s < 2; ++s) {
            union { v16bf v; bf16x8 h2[2]; } u;
            u.h2[0] = *(const bf16x8*)(qrow + s * 32 + lhalf * 8);
            u.h2[1] = *(const bf16x8*)(qrow + s * 32 + lhalf * 8 + 16);
            qf[s] = u.v;
        }
    }

    v8f oacc[4] = {};
    float m[8], l[8];
#pragma unroll
    for (int i = 0; i < 8; ++i) { m[i] = -1e30f; l[i] = 0.f; }

    const int ktiles = qblk * 4 + 4;   // keys up to end of this query block
    for (int kt = 0; kt < ktiles; ++kt) {
        __syncthreads();
        // stage K tile [32][64], 8 threads per key row, 16B each
        {
            int kr = tid >> 3, dseg = (tid & 7) * 8;
            const unsigned short* src = Kb + headoff + (size_t)(kt * 32 + kr) * 1024 + dseg;
            *(uint4*)&sK[kr][dseg] = *(const uint4*)src;
        }
        // stage V tile transposed -> sVT[d][key]
        {
            int vr = tid >> 3, dseg = (tid & 7) * 8;
            const unsigned short* src = Vb + headoff + (size_t)(kt * 32 + vr) * 1024 + dseg;
            unsigned short tmp[8];
            *(uint4*)tmp = *(const uint4*)src;
#pragma unroll
            for (int e = 0; e < 8; ++e) sVT[dseg + e][vr] = tmp[e];
        }
        __syncthreads();

        // S = Q K^T for two 16-key subtiles (contraction hd=64 in two steps)
        float sv[2][8];
#pragma unroll
        for (int n = 0; n < 2; ++n) {
            v8f sacc = {};
#pragma unroll
            for (int s = 0; s < 2; ++s) {
                union { v16bf v; bf16x8 h2[2]; } uk;
                uk.h2[0] = *(const bf16x8*)&sK[n * 16 + lrow][s * 32 + lhalf * 8];
                uk.h2[1] = *(const bf16x8*)&sK[n * 16 + lrow][s * 32 + lhalf * 8 + 16];
                v16bf kf = uk.v;
                sacc = __builtin_amdgcn_wmma_f32_16x16x32_bf16(
                    false, qf[s], false, kf, (short)0, sacc, false, false);
            }
            int kg = kt * 32 + n * 16 + lrow;   // key column for this lane
#pragma unroll
            for (int i = 0; i < 8; ++i) {
                int qg = qbase + i + lhalf * 8;
                float val = sacc[i] * 0.125f;   // SCALE = 1/sqrt(64)
                sv[n][i] = (kg > qg) ? -1e30f : val;
            }
        }

        // online softmax; row r lives in VGPR i across the lane's 16-lane half
        float p[2][8];
#pragma unroll
        for (int i = 0; i < 8; ++i) {
            float t = fmaxf(sv[0][i], sv[1][i]);
            t = fmaxf(t, __shfl_xor(t, 1, 32));
            t = fmaxf(t, __shfl_xor(t, 2, 32));
            t = fmaxf(t, __shfl_xor(t, 4, 32));
            t = fmaxf(t, __shfl_xor(t, 8, 32));
            float mn = fmaxf(m[i], t);
            p[0][i] = __expf(sv[0][i] - mn);
            p[1][i] = __expf(sv[1][i] - mn);
            float rs = p[0][i] + p[1][i];
            rs += __shfl_xor(rs, 1, 32);
            rs += __shfl_xor(rs, 2, 32);
            rs += __shfl_xor(rs, 4, 32);
            rs += __shfl_xor(rs, 8, 32);
            float sc = __expf(m[i] - mn);
            l[i] = l[i] * sc + rs;
            m[i] = mn;
#pragma unroll
            for (int j = 0; j < 4; ++j) oacc[j][i] = oacc[j][i] * sc;
        }

        // P (C-layout) -> LDS -> A-fragment layout
#pragma unroll
        for (int n = 0; n < 2; ++n)
#pragma unroll
            for (int i = 0; i < 8; ++i)
                sP[wave][i + lhalf * 8][n * 16 + lrow] = f32_to_bf16(p[n][i]);

        union { v16bf v; bf16x8 h2[2]; } up;
        up.h2[0] = *(const bf16x8*)&sP[wave][lrow][lhalf * 8];
        up.h2[1] = *(const bf16x8*)&sP[wave][lrow][lhalf * 8 + 16];
        v16bf pf = up.v;

        // O += P * V  (B-frag: column = d, K = key; sVT rows are contiguous)
#pragma unroll
        for (int j = 0; j < 4; ++j) {
            union { v16bf v; bf16x8 h2[2]; } uv;
            uv.h2[0] = *(const bf16x8*)&sVT[j * 16 + lrow][lhalf * 8];
            uv.h2[1] = *(const bf16x8*)&sVT[j * 16 + lrow][lhalf * 8 + 16];
            v16bf vf = uv.v;
            oacc[j] = __builtin_amdgcn_wmma_f32_16x16x32_bf16(
                false, pf, false, vf, (short)0, oacc[j], false, false);
        }
    }

    // finalize: divide by l, store bf16 to [B*T][1024]
#pragma unroll
    for (int j = 0; j < 4; ++j) {
#pragma unroll
        for (int i = 0; i < 8; ++i) {
            float inv = (l[i] > 0.f) ? 1.0f / l[i] : 0.f;
            float v = oacc[j][i] * inv;
            int row = qbase + i + lhalf * 8;
            int col = h * 64 + j * 16 + lrow;
            O[(size_t)(b * T + row) * 1024 + col] = f32_to_bf16(v);
        }
    }
}

// ---------------------------------------------------------------------------
extern "C" void kernel_launch(void* const* d_in, const int* in_sizes, int n_in,
                              void* d_out, int out_size, void* d_ws, size_t ws_size,
                              hipStream_t stream) {
    const float* x  = (const float*)d_in[0];
    const float* Wq = (const float*)d_in[1];
    const float* bq = (const float*)d_in[2];
    const float* Wk = (const float*)d_in[3];
    const float* bk = (const float*)d_in[4];
    const float* Wv = (const float*)d_in[5];
    const float* bv = (const float*)d_in[6];
    const float* Wo = (const float*)d_in[7];
    const float* bo = (const float*)d_in[8];
    float* out = (float*)d_out;

    const int B = 4, T = 2048, D = 1024;
    const int M = B * T;  // 8192

    char* ws = (char*)d_ws;
    unsigned short* xb  = (unsigned short*)ws;  ws += (size_t)M * D * 2;
    unsigned short* WqT = (unsigned short*)ws;  ws += (size_t)D * D * 2;
    unsigned short* WkT = (unsigned short*)ws;  ws += (size_t)D * D * 2;
    unsigned short* WvT = (unsigned short*)ws;  ws += (size_t)D * D * 2;
    unsigned short* WoT = (unsigned short*)ws;  ws += (size_t)D * D * 2;
    unsigned short* Qb  = (unsigned short*)ws;  ws += (size_t)M * D * 2;
    unsigned short* Kbf = (unsigned short*)ws;  ws += (size_t)M * D * 2;
    unsigned short* Vbf = (unsigned short*)ws;  ws += (size_t)M * D * 2;
    unsigned short* Ab  = (unsigned short*)ws;  ws += (size_t)M * D * 2;

    cvt_f32_bf16<<<2048, 256, 0, stream>>>(x, xb, M * D);
    tcvt_f32_bf16<<<1024, 256, 0, stream>>>(Wq, WqT, D, D);
    tcvt_f32_bf16<<<1024, 256, 0, stream>>>(Wk, WkT, D, D);
    tcvt_f32_bf16<<<1024, 256, 0, stream>>>(Wv, WvT, D, D);
    tcvt_f32_bf16<<<1024, 256, 0, stream>>>(Wo, WoT, D, D);

    dim3 gg(D / 128, M / 128);
    gemm_wmma<true><<<gg, 256, 0, stream>>>(xb, WqT, bq, Qb, M, D, D);
    gemm_wmma<true><<<gg, 256, 0, stream>>>(xb, WkT, bk, Kbf, M, D, D);
    gemm_wmma<true><<<gg, 256, 0, stream>>>(xb, WvT, bv, Vbf, M, D, D);

    attn_wmma<<<dim3(T / 128, B * 16), 256, 0, stream>>>(Qb, Kbf, Vbf, Ab, T);

    gemm_wmma<false><<<gg, 256, 0, stream>>>(Ab, WoT, bo, out, M, D, D);
}
